// Q2A_Function_36197984370923
// MI455X (gfx1250) — compile-verified, matrix operand order checked by reading, not run on hardware
//
#include <hip/hip_runtime.h>

// ---------------------------------------------------------------------------
// Types for CDNA5 WMMA (gfx1250, wave32)
// ---------------------------------------------------------------------------
typedef __bf16 bf16;
typedef __attribute__((__ext_vector_type__(16))) __bf16 v16bf;
typedef __attribute__((__ext_vector_type__(8)))  float  v8f;
typedef __attribute__((__ext_vector_type__(4)))  float  f32x4;

__device__ __forceinline__ bf16 to_bf16(float f) {
  union { float f; unsigned u; } a; a.f = f;
  unsigned r = a.u + 0x7FFFu + ((a.u >> 16) & 1u);   // round-to-nearest-even
  unsigned short hs = (unsigned short)(r >> 16);
  union { unsigned short s; bf16 b; } c; c.s = hs; return c.b;
}

__device__ __forceinline__ float q2a_sigmoid(float x) { return 1.f / (1.f + expf(-x)); }

template <int ACT>
__device__ __forceinline__ float act_fn(float x) {
  if (ACT == 1) return x > 0.f ? x : 0.f;     // relu
  if (ACT == 2) return tanhf(x);              // tanh
  if (ACT == 3) return q2a_sigmoid(x);        // sigmoid
  return x;
}

// ---------------------------------------------------------------------------
// WMMA GEMM:  C[M,N] = act( A[M,K](bf16, lda) @ Bt[N,K]^T (bf16) + bias[N] )
//
// Block: 256 thr = 8 waves arranged 2(M) x 4(N); each wave owns a 32x64 tile
// (2 M-tiles x 4 N-tiles = 8 accumulators). Block tile = 64 x 256.
// Requires M%64==0, N%256==0, K%64==0 (all shapes here satisfy this).
// Grid sized exactly -> EXEC all-ones for every v_wmma.
//
// Double-buffered K loop (step 64, two 32-wide phases, ping-pong fragment
// sets, no copies): loads for the next phase are issued before the current
// phase's 8 WMMAs, so the hardware only needs s_wait_loadcnt <= 12 instead of
// draining to 0 before every WMMA -> 12 b128 loads stay in flight under the
// matrix ops.
//
// Per ISA 16-bit A layout: lane m=lane%16, elements e: K=(e/8)*16+(lane/16)*8+e%8
// -> two contiguous 8-half (16B) loads per fragment.
// ---------------------------------------------------------------------------
template <int ACT>
__global__ __launch_bounds__(256)
void q2a_gemm(const bf16* __restrict__ A, int lda,
              const bf16* __restrict__ Bt,
              const float* __restrict__ bias,
              float* __restrict__ C, int ldc,
              bf16* __restrict__ Cbf, int ldcb,
              int N, int K)
{
  const int wave = threadIdx.x >> 5;
  const int lane = threadIdx.x & 31;
  const int m16  = lane & 15;
  const int kh   = (lane >> 4) << 3;   // 0 or 8 : K sub-half select
  const int mhi  = (lane >> 4) << 3;   // C/D layout: M = r + (lane/16)*8
  const int row0 = blockIdx.x * 64  + (wave & 1)  * 32;
  const int col0 = blockIdx.y * 256 + (wave >> 1) * 64;

  v8f acc[2][4];
#pragma unroll
  for (int mt = 0; mt < 2; ++mt)
#pragma unroll
    for (int nt = 0; nt < 4; ++nt)
#pragma unroll
      for (int r = 0; r < 8; ++r) acc[mt][nt][r] = 0.f;

  const bf16* arow0 = A + (size_t)(row0 + m16) * lda + kh;
  const bf16* arow1 = A + (size_t)(row0 + 16 + m16) * lda + kh;
  const bf16* brow[4];
#pragma unroll
  for (int nt = 0; nt < 4; ++nt)
    brow[nt] = Bt + (size_t)(col0 + nt * 16 + m16) * K + kh;

  union Fr { f32x4 q[2]; v16bf v; };
  Fr aA[2], bA[4], aB[2], bB[4];

  auto ldfrag = [](Fr& f, const bf16* p) {
    f.q[0] = *(const f32x4*)(p);
    f.q[1] = *(const f32x4*)(p + 16);
  };

  // preload phase-A fragments (kb = 0)
  ldfrag(aA[0], arow0); ldfrag(aA[1], arow1);
#pragma unroll
  for (int nt = 0; nt < 4; ++nt) ldfrag(bA[nt], brow[nt]);

  for (int kb = 0; kb < K; kb += 64) {
    const int k1 = kb + 32;
    // issue phase-B loads before consuming phase-A
    ldfrag(aB[0], arow0 + k1); ldfrag(aB[1], arow1 + k1);
#pragma unroll
    for (int nt = 0; nt < 4; ++nt) ldfrag(bB[nt], brow[nt] + k1);

#pragma unroll
    for (int nt = 0; nt < 4; ++nt) {
      acc[0][nt] = __builtin_amdgcn_wmma_f32_16x16x32_bf16(
          false, aA[0].v, false, bA[nt].v, (short)0, acc[0][nt], false, false);
      acc[1][nt] = __builtin_amdgcn_wmma_f32_16x16x32_bf16(
          false, aA[1].v, false, bA[nt].v, (short)0, acc[1][nt], false, false);
    }

    const int k2 = kb + 64;
    if (k2 < K) {   // uniform branch; refill phase-A for next iteration
      ldfrag(aA[0], arow0 + k2); ldfrag(aA[1], arow1 + k2);
#pragma unroll
      for (int nt = 0; nt < 4; ++nt) ldfrag(bA[nt], brow[nt] + k2);
    }

#pragma unroll
    for (int nt = 0; nt < 4; ++nt) {
      acc[0][nt] = __builtin_amdgcn_wmma_f32_16x16x32_bf16(
          false, aB[0].v, false, bB[nt].v, (short)0, acc[0][nt], false, false);
      acc[1][nt] = __builtin_amdgcn_wmma_f32_16x16x32_bf16(
          false, aB[1].v, false, bB[nt].v, (short)0, acc[1][nt], false, false);
    }
  }

#pragma unroll
  for (int mt = 0; mt < 2; ++mt) {
#pragma unroll
    for (int nt = 0; nt < 4; ++nt) {
      const int n = col0 + nt * 16 + m16;
      const float bv = bias ? bias[n] : 0.f;
#pragma unroll
      for (int r = 0; r < 8; ++r) {
        const int m = row0 + mt * 16 + mhi + r;
        float v = act_fn<ACT>(acc[mt][nt][r] + bv);
        if (C)   C[(size_t)m * ldc + n] = v;
        if (Cbf) Cbf[(size_t)m * ldcb + n] = to_bf16(v);
      }
    }
  }
}

// ---------------------------------------------------------------------------
// LayerNorm rows: Y(bf16) = (X - mean)/sqrt(var+1e-5)*g + b.  One block/row.
// ---------------------------------------------------------------------------
__global__ __launch_bounds__(256)
void q2a_ln(const float* __restrict__ X, const float* __restrict__ g,
            const float* __restrict__ b, bf16* __restrict__ Y, int Dd)
{
  const int row = blockIdx.x;
  const float* x = X + (size_t)row * Dd;
  bf16* y = Y + (size_t)row * Dd;
  __shared__ float red[256];
  float s = 0.f, s2 = 0.f;
  for (int d = threadIdx.x; d < Dd; d += 256) { float v = x[d]; s += v; s2 += v * v; }
  red[threadIdx.x] = s; __syncthreads();
  for (int o = 128; o > 0; o >>= 1) { if (threadIdx.x < o) red[threadIdx.x] += red[threadIdx.x + o]; __syncthreads(); }
  const float mean = red[0] / Dd;
  __syncthreads();
  red[threadIdx.x] = s2; __syncthreads();
  for (int o = 128; o > 0; o >>= 1) { if (threadIdx.x < o) red[threadIdx.x] += red[threadIdx.x + o]; __syncthreads(); }
  const float var = red[0] / Dd - mean * mean;
  const float inv = rsqrtf(var + 1e-5f);
  for (int d = threadIdx.x; d < Dd; d += 256)
    y[d] = to_bf16((x[d] - mean) * inv * g[d] + b[d]);
}

// proj head: logits[row] = dot( LN(X_row; g,b), w ) + c2[0]
__global__ __launch_bounds__(256)
void q2a_proj_ln_dot(const float* __restrict__ X, const float* __restrict__ g,
                     const float* __restrict__ b, const float* __restrict__ w,
                     const float* __restrict__ c2, float* __restrict__ logits, int Dd)
{
  const int row = blockIdx.x;
  const float* x = X + (size_t)row * Dd;
  __shared__ float red[256];
  float s = 0.f, s2 = 0.f;
  for (int d = threadIdx.x; d < Dd; d += 256) { float v = x[d]; s += v; s2 += v * v; }
  red[threadIdx.x] = s; __syncthreads();
  for (int o = 128; o > 0; o >>= 1) { if (threadIdx.x < o) red[threadIdx.x] += red[threadIdx.x + o]; __syncthreads(); }
  const float mean = red[0] / Dd;
  __syncthreads();
  red[threadIdx.x] = s2; __syncthreads();
  for (int o = 128; o > 0; o >>= 1) { if (threadIdx.x < o) red[threadIdx.x] += red[threadIdx.x + o]; __syncthreads(); }
  const float var = red[0] / Dd - mean * mean;
  const float inv = rsqrtf(var + 1e-5f);
  __syncthreads();
  float acc = 0.f;
  for (int d = threadIdx.x; d < Dd; d += 256)
    acc += ((x[d] - mean) * inv * g[d] + b[d]) * w[d];
  red[threadIdx.x] = acc; __syncthreads();
  for (int o = 128; o > 0; o >>= 1) { if (threadIdx.x < o) red[threadIdx.x] += red[threadIdx.x + o]; __syncthreads(); }
  if (threadIdx.x == 0) logits[row] = red[0] + c2[0];
}

// Row softmax (one wave/row): Y = exp(x-max)/(sum+eps). In-place safe.
__global__ void q2a_rowsoftmax(const float* __restrict__ X, float* __restrict__ Y,
                               int L, float eps)
{
  const int b = blockIdx.x, lane = threadIdx.x;
  const float* x = X + (size_t)b * L;
  float* y = Y + (size_t)b * L;
  float mx = -1e30f;
  for (int i = lane; i < L; i += 32) mx = fmaxf(mx, x[i]);
  for (int o = 16; o > 0; o >>= 1) mx = fmaxf(mx, __shfl_xor(mx, o, 32));
  float s = 0.f;
  for (int i = lane; i < L; i += 32) { float e = expf(x[i] - mx); y[i] = e; s += e; }
  for (int o = 16; o > 0; o >>= 1) s += __shfl_xor(s, o, 32);
  const float inv = 1.f / (s + eps);
  for (int i = lane; i < L; i += 32) y[i] *= inv;
}

__global__ void q2a_attn_dot(const float* __restrict__ qq, const float* __restrict__ kk,
                             float* __restrict__ alpha, int V, int Dd)
{
  const int bv = blockIdx.x, lane = threadIdx.x;
  const float* qr = qq + (size_t)(bv / V) * Dd;
  const float* kr = kk + (size_t)bv * Dd;
  float s = 0.f;
  for (int d = lane; d < Dd; d += 32) s += qr[d] * kr[d];
  for (int o = 16; o > 0; o >>= 1) s += __shfl_xor(s, o, 32);
  if (lane == 0) alpha[bv] = s;
}

__global__ void q2a_pool_para(const float* __restrict__ w, const float* __restrict__ pm,
                              const float* __restrict__ te, float* __restrict__ out,
                              int P, int Dd, int n)
{
  int i = blockIdx.x * blockDim.x + threadIdx.x; if (i >= n) return;
  const int b = i / Dd, d = i % Dd;
  float s = 0.f;
  for (int p = 0; p < P; ++p)
    s += w[b * P + p] * (pm[((size_t)b * P + p) * Dd + d] + te[(size_t)p * Dd + d]);
  out[i] = s;
}

__global__ void q2a_pool_video(const float* __restrict__ a, const float* __restrict__ vid,
                               float* __restrict__ out, int V, int Dd, int n)
{
  int i = blockIdx.x * blockDim.x + threadIdx.x; if (i >= n) return;
  const int b = i / Dd, d = i % Dd;
  float s = 0.f;
  for (int v = 0; v < V; ++v)
    s += a[b * V + v] * vid[((size_t)b * V + v) * Dd + d];
  out[i] = s;
}

__global__ void q2a_add_te(float* __restrict__ x, const float* __restrict__ te,
                           int V, int Dd, int n)
{
  int i = blockIdx.x * blockDim.x + threadIdx.x; if (i >= n) return;
  const int v = (i / Dd) % V;
  x[i] += te[(size_t)v * Dd + i % Dd];
}

__global__ void q2a_cvt_bf16(const float* __restrict__ x, bf16* __restrict__ y, int n)
{
  int i = blockIdx.x * blockDim.x + threadIdx.x;
  if (i < n) y[i] = to_bf16(x[i]);
}

// fp32 W[K,N] -> bf16 Wt[N,K] via 32x32 LDS tile transpose.
// Both the fp32 reads and the bf16 writes are coalesced; +1 pad kills bank
// conflicts. Requires K%32==0, N%32==0 (true for all weights here).
__global__ __launch_bounds__(256)
void q2a_cvtT(const float* __restrict__ W, bf16* __restrict__ Wt, int K, int N)
{
  __shared__ float tile[32][33];
  const int k0 = blockIdx.x * 32, n0 = blockIdx.y * 32;
  const int tx = threadIdx.x & 31, ty = threadIdx.x >> 5;  // 8 rows / pass
#pragma unroll
  for (int r = ty; r < 32; r += 8)
    tile[r][tx] = W[(size_t)(k0 + r) * N + (n0 + tx)];
  __syncthreads();
#pragma unroll
  for (int r = ty; r < 32; r += 8)
    Wt[(size_t)(n0 + r) * K + (k0 + tx)] = to_bf16(tile[tx][r]);
}

// a_texts/a_buttons[:, s] -> contiguous [B*A, D]
__global__ void q2a_slice_step(const float* __restrict__ src, float* __restrict__ dst,
                               int s, int S, int A, int Dd, int n)
{
  int i = blockIdx.x * blockDim.x + threadIdx.x; if (i >= n) return;
  const int d = i % Dd, row = i / Dd, a = row % A, b = row / A;
  dst[i] = src[(((size_t)b * S + s) * A + a) * Dd + d];
}

// x4[:, 0:3D] = [video_seg | text_seg | q] broadcast per (b,a); filled once.
__global__ void q2a_fill_x4(const float* __restrict__ vs, const float* __restrict__ ts,
                            const float* __restrict__ q, float* __restrict__ x4,
                            int A, int Dd, int n)
{
  int i = blockIdx.x * blockDim.x + threadIdx.x; if (i >= n) return;
  const int c = i % (3 * Dd), row = i / (3 * Dd), b = row / A;
  const int seg = c / Dd, d = c % Dd;
  const float* src = (seg == 0) ? vs : (seg == 1) ? ts : q;
  x4[(size_t)row * 4 * Dd + c] = src[(size_t)b * Dd + d];
}

// x4[:, 3D:4D] = z*tanh1 + (1-z)*tanh2  (z already sigmoid'd, t1/t2 already tanh'd)
__global__ void q2a_gate_combine(const float* __restrict__ z, const float* __restrict__ t1,
                                 const float* __restrict__ t2, float* __restrict__ x4,
                                 int Dd, int n)
{
  int i = blockIdx.x * blockDim.x + threadIdx.x; if (i >= n) return;
  const int row = i / Dd, d = i % Dd;
  x4[(size_t)row * 4 * Dd + 3 * Dd + d] = z[i] * t1[i] + (1.f - z[i]) * t2[i];
}

__global__ void q2a_gru(const float* __restrict__ gi, const float* __restrict__ gh,
                        const float* __restrict__ st, float* __restrict__ states,
                        int A, int Hh, int n)
{
  int i = blockIdx.x * blockDim.x + threadIdx.x; if (i >= n) return;
  const int row = i / Hh, col = i % Hh, b = row / A;
  const size_t gr = (size_t)row * 3 * Hh, gb = (size_t)b * 3 * Hh;
  const float ir = gi[gr + col], iz = gi[gr + Hh + col], in = gi[gr + 2 * Hh + col];
  const float hr = gh[gb + col], hz = gh[gb + Hh + col], hn = gh[gb + 2 * Hh + col];
  const float h = st[(size_t)b * Hh + col];
  const float r = q2a_sigmoid(ir + hr);
  const float z = q2a_sigmoid(iz + hz);
  const float nn = tanhf(in + r * hn);
  states[i] = (1.f - z) * nn + z * h;
}

__global__ void q2a_state_init(const float* __restrict__ s0, float* __restrict__ st,
                               bf16* __restrict__ stbf, int Dd, int n)
{
  int i = blockIdx.x * blockDim.x + threadIdx.x; if (i >= n) return;
  const float v = s0[i % Dd];
  st[i] = v; stbf[i] = to_bf16(v);
}

__global__ void q2a_state_gather(const float* __restrict__ inp, const int* __restrict__ lab,
                                 int s, int S, int A, int Dd,
                                 float* __restrict__ st, bf16* __restrict__ stbf, int n)
{
  int i = blockIdx.x * blockDim.x + threadIdx.x; if (i >= n) return;
  const int b = i / Dd, d = i % Dd;
  const int l = lab[b * S + s];
  const float v = inp[((size_t)b * A + l) * Dd + d];
  st[i] = v; stbf[i] = to_bf16(v);
}

__global__ void q2a_loss(const float* __restrict__ logits, const int* __restrict__ lab,
                         int s, int S, int A, float scale, float* __restrict__ out)
{
  const int b = blockIdx.x, lane = threadIdx.x;
  float v = (lane < A) ? logits[b * A + lane] : -1e30f;
  float mx = v;
  for (int o = 16; o > 0; o >>= 1) mx = fmaxf(mx, __shfl_xor(mx, o, 32));
  float e = (lane < A) ? expf(logits[b * A + lane] - mx) : 0.f;
  float sum = e;
  for (int o = 16; o > 0; o >>= 1) sum += __shfl_xor(sum, o, 32);
  if (lane == 0) {
    const int l = lab[b * S + s];
    const float lp = (logits[b * A + l] - mx) - logf(sum);
    atomicAdd(out, -lp * scale);
  }
}

__global__ void q2a_zero(float* out) { out[0] = 0.f; }

// ---------------------------------------------------------------------------
// Host orchestration
// ---------------------------------------------------------------------------
static void launch_gemm(int act, const bf16* A, int lda, const bf16* Bt,
                        const float* bias, float* C, int ldc, bf16* Cb, int ldcb,
                        int M, int N, int K, hipStream_t st)
{
  dim3 grid(M / 64, N / 256), blk(256);
  switch (act) {
    case 1: q2a_gemm<1><<<grid, blk, 0, st>>>(A, lda, Bt, bias, C, ldc, Cb, ldcb, N, K); break;
    case 2: q2a_gemm<2><<<grid, blk, 0, st>>>(A, lda, Bt, bias, C, ldc, Cb, ldcb, N, K); break;
    case 3: q2a_gemm<3><<<grid, blk, 0, st>>>(A, lda, Bt, bias, C, ldc, Cb, ldcb, N, K); break;
    default: q2a_gemm<0><<<grid, blk, 0, st>>>(A, lda, Bt, bias, C, ldc, Cb, ldcb, N, K); break;
  }
}

extern "C" void kernel_launch(void* const* d_in, const int* in_sizes, int n_in,
                              void* d_out, int out_size, void* d_ws, size_t ws_size,
                              hipStream_t stream)
{
  (void)in_sizes; (void)n_in; (void)out_size; (void)ws_size;
  constexpr int B = 128, V = 50, P = 32, S = 12, A = 16, D = 768, H = 768;
  constexpr int D4 = 4 * D, H3 = 3 * H;

  const float* F[55];
  for (int i = 0; i < 55; ++i) F[i] = (const float*)d_in[i];
  const int* labels = (const int*)d_in[6];

  // mlpv / mlpt / mlppre / proj params (g1,b1,w1,c1,g2,b2,w2,c2)
  const float *mv_g1=F[7], *mv_b1=F[8], *mv_w1=F[9],  *mv_c1=F[10], *mv_g2=F[11], *mv_b2=F[12], *mv_w2=F[13], *mv_c2=F[14];
  const float *mt_g1=F[15],*mt_b1=F[16],*mt_w1=F[17], *mt_c1=F[18], *mt_g2=F[19], *mt_b2=F[20], *mt_w2=F[21], *mt_c2=F[22];
  const float *pr_g1=F[23],*pr_b1=F[24],*pr_w1=F[25], *pr_c1=F[26], *pr_g2=F[27], *pr_b2=F[28], *pr_w2=F[29], *pr_c2=F[30];
  const float *pj_g1=F[31],*pj_b1=F[32],*pj_w1=F[33], *pj_c1=F[34], *pj_g2=F[35], *pj_b2=F[36], *pj_w2=F[37], *pj_c2=F[38];
  const float *gw1=F[39], *gc1=F[40], *gw2=F[41], *gc2=F[42], *gwa=F[43], *gca=F[44];
  const float *wk=F[45],  *bk=F[46],  *wq=F[47],  *bq=F[48];
  const float *wih=F[49], *whh=F[50], *bih=F[51], *bhh=F[52];
  const float *timeemb=F[53], *state0=F[54];

  // ------- workspace arena -------
  size_t off = 0;
  auto allocb = [&](size_t bytes) -> char* {
    char* p = (char*)d_ws + off; off += (bytes + 255) & ~(size_t)255; return p;
  };
  auto allocf = [&](size_t n) -> float* { return (float*)allocb(n * sizeof(float)); };
  auto alloch = [&](size_t n) -> bf16*  { return (bf16*)allocb(n * sizeof(bf16)); };

  // bf16 transposed weights
  bf16 *Wmt1 = alloch((size_t)D * D),  *Wmt2 = alloch((size_t)D * D);
  bf16 *Wmv1 = alloch((size_t)D * D),  *Wmv2 = alloch((size_t)D * D);
  bf16 *Wpr1 = alloch((size_t)D4 * D4), *Wpr2 = alloch((size_t)D4 * H);
  bf16 *Wpj1 = alloch((size_t)D * D);
  bf16 *Wwk  = alloch((size_t)D * D),  *Wwq  = alloch((size_t)D * D);
  bf16 *Wg1  = alloch((size_t)D * D),  *Wg2  = alloch((size_t)D * D);
  bf16 *Wga  = alloch((size_t)2 * D * D);
  bf16 *Wih  = alloch((size_t)H * H3), *Whh  = alloch((size_t)H * H3);

  const int M = B * A;  // 2048
  float *big0   = allocf((size_t)B * V * D);     // vid (persists through attn)
  float *big1   = allocf((size_t)B * V * D);     // para_out, later kk
  bf16  *vid_bf = alloch((size_t)B * V * D);
  bf16  *lnbuf  = alloch((size_t)M * D4);        // LN outputs (max 2048x3072)
  float *hbuf   = allocf((size_t)M * D4);        // relu intermediates
  float *x4     = allocf((size_t)M * D4);        // [vseg|tseg|q|btn_gated]
  float *gi     = allocf((size_t)M * H3);
  float *gh     = allocf((size_t)B * H3);
  bf16  *cat_bf = alloch((size_t)M * 2 * D);     // [btn | t] bf16
  float *zb   = allocf((size_t)M * D);
  float *t1b  = allocf((size_t)M * D);
  float *t2b  = allocf((size_t)M * D);
  float *sbuf = allocf((size_t)M * D);
  float *inp  = allocf((size_t)M * D);  bf16 *inp_bf = alloch((size_t)M * D);
  float *states = allocf((size_t)M * D);
  float *projh  = allocf((size_t)M * D);
  float *logits = allocf((size_t)M);
  float *qv   = allocf((size_t)B * D);  bf16 *q_bf = alloch((size_t)B * D);
  float *qq   = allocf((size_t)B * D);
  float *tseg = allocf((size_t)B * D);
  float *vseg = allocf((size_t)B * D);
  float *sw   = allocf((size_t)B * P);
  float *alpha = allocf((size_t)B * V);
  float *st   = allocf((size_t)B * D);  bf16 *st_bf = alloch((size_t)B * D);

  auto EW = [](int n) { return dim3((n + 255) / 256); };
  auto cvtT = [&](const float* W, bf16* Wt, int K, int N) {
    q2a_cvtT<<<dim3(K / 32, N / 32), 256, 0, stream>>>(W, Wt, K, N);
  };
  auto LN = [&](const float* X, const float* g, const float* b, bf16* Y, int rows, int Dd) {
    q2a_ln<<<rows, 256, 0, stream>>>(X, g, b, Y, Dd);
  };
  auto GEMM = [&](int act, const bf16* Aa, int lda, const bf16* Bt, const float* bias,
                  float* C, int ldc, bf16* Cb, int ldcb, int Mm, int Nn, int Kk) {
    launch_gemm(act, Aa, lda, Bt, bias, C, ldc, Cb, ldcb, Mm, Nn, Kk, stream);
  };

  // ------- weight conversion (fp32 [K,N] -> bf16 [N,K]) -------
  cvtT(mt_w1, Wmt1, D, D);   cvtT(mt_w2, Wmt2, D, D);
  cvtT(mv_w1, Wmv1, D, D);   cvtT(mv_w2, Wmv2, D, D);
  cvtT(pr_w1, Wpr1, D4, D4); cvtT(pr_w2, Wpr2, D4, H);
  cvtT(pj_w1, Wpj1, D, D);
  cvtT(wk, Wwk, D, D);       cvtT(wq, Wwq, D, D);
  cvtT(gw1, Wg1, D, D);      cvtT(gw2, Wg2, D, D);  cvtT(gwa, Wga, 2 * D, D);
  cvtT(wih, Wih, H, H3);     cvtT(whh, Whh, H, H3);

  q2a_zero<<<1, 1, 0, stream>>>((float*)d_out);

  // ------- stage B: per-batch constants -------
  // q = mlpt(question)
  LN(F[2], mt_g1, mt_b1, lnbuf, B, D);
  GEMM(1, lnbuf, D, Wmt1, mt_c1, hbuf, D, nullptr, 0, B, D, D);
  LN(hbuf, mt_g2, mt_b2, lnbuf, B, D);
  GEMM(0, lnbuf, D, Wmt2, mt_c2, qv, D, q_bf, D, B, D, D);

  // text_seg = softmax(paras_score) . (mlpt(para) + timeemb[:P])
  q2a_rowsoftmax<<<B, 32, 0, stream>>>(F[3], sw, P, 0.f);
  LN(F[1], mt_g1, mt_b1, lnbuf, B * P, D);
  GEMM(1, lnbuf, D, Wmt1, mt_c1, hbuf, D, nullptr, 0, B * P, D, D);
  LN(hbuf, mt_g2, mt_b2, lnbuf, B * P, D);
  GEMM(0, lnbuf, D, Wmt2, mt_c2, big1, D, nullptr, 0, B * P, D, D);
  q2a_pool_para<<<EW(B * D), 256, 0, stream>>>(sw, big1, timeemb, tseg, P, D, B * D);

  // vid = mlpv(video) + timeemb[:V]; attention pooling -> video_seg
  LN(F[0], mv_g1, mv_b1, lnbuf, B * V, D);
  GEMM(1, lnbuf, D, Wmv1, mv_c1, hbuf, D, nullptr, 0, B * V, D, D);
  LN(hbuf, mv_g2, mv_b2, lnbuf, B * V, D);
  GEMM(0, lnbuf, D, Wmv2, mv_c2, big0, D, nullptr, 0, B * V, D, D);
  q2a_add_te<<<EW(B * V * D), 256, 0, stream>>>(big0, timeemb, V, D, B * V * D);
  q2a_cvt_bf16<<<EW(B * V * D), 256, 0, stream>>>(big0, vid_bf, B * V * D);
  GEMM(0, vid_bf, D, Wwk, bk, big1, D, nullptr, 0, B * V, D, D);   // kk
  GEMM(0, q_bf, D, Wwq, bq, qq, D, nullptr, 0, B, D, D);           // qq
  q2a_attn_dot<<<B * V, 32, 0, stream>>>(qq, big1, alpha, V, D);
  q2a_rowsoftmax<<<B, 32, 0, stream>>>(alpha, alpha, V, 1e-8f);
  q2a_pool_video<<<EW(B * D), 256, 0, stream>>>(alpha, big0, vseg, V, D, B * D);

  // static 3/4 of the x4 concat, initial GRU state
  q2a_fill_x4<<<EW(M * 3 * D), 256, 0, stream>>>(vseg, tseg, qv, x4, A, D, M * 3 * D);
  q2a_state_init<<<EW(B * D), 256, 0, stream>>>(state0, st, st_bf, D, B * D);

  // ------- stage C: steps (state recursion is just a gather of inp[label]) --
  const float scale = 1.0f / (float)(S * B);
  for (int s = 0; s < S; ++s) {
    // t = mlpt(a_texts[:,s]) -> bf16 into cat[:, D:2D]
    q2a_slice_step<<<EW(M * D), 256, 0, stream>>>(F[4], sbuf, s, S, A, D, M * D);
    LN(sbuf, mt_g1, mt_b1, lnbuf, M, D);
    GEMM(1, lnbuf, D, Wmt1, mt_c1, hbuf, D, nullptr, 0, M, D, D);
    LN(hbuf, mt_g2, mt_b2, lnbuf, M, D);
    GEMM(0, lnbuf, D, Wmt2, mt_c2, nullptr, 0, cat_bf + D, 2 * D, M, D, D);

    // btn = mlpv(a_buttons[:,s]) -> bf16 into cat[:, 0:D]
    q2a_slice_step<<<EW(M * D), 256, 0, stream>>>(F[5], sbuf, s, S, A, D, M * D);
    LN(sbuf, mv_g1, mv_b1, lnbuf, M, D);
    GEMM(1, lnbuf, D, Wmv1, mv_c1, hbuf, D, nullptr, 0, M, D, D);
    LN(hbuf, mv_g2, mv_b2, lnbuf, M, D);
    GEMM(0, lnbuf, D, Wmv2, mv_c2, nullptr, 0, cat_bf, 2 * D, M, D, D);

    // gate: z = sigmoid([btn,t]@gwa+gca); btn_g = z*tanh(btn@gw1+gc1)+(1-z)*tanh(t@gw2+gc2)
    GEMM(3, cat_bf,     2 * D, Wga, gca, zb,  D, nullptr, 0, M, D, 2 * D);
    GEMM(2, cat_bf,     2 * D, Wg1, gc1, t1b, D, nullptr, 0, M, D, D);
    GEMM(2, cat_bf + D, 2 * D, Wg2, gc2, t2b, D, nullptr, 0, M, D, D);
    q2a_gate_combine<<<EW(M * D), 256, 0, stream>>>(zb, t1b, t2b, x4, D, M * D);

    // inp = mlppre(x4)
    LN(x4, pr_g1, pr_b1, lnbuf, M, D4);
    GEMM(1, lnbuf, D4, Wpr1, pr_c1, hbuf, D4, nullptr, 0, M, D4, D4);
    LN(hbuf, pr_g2, pr_b2, lnbuf, M, D4);
    GEMM(0, lnbuf, D4, Wpr2, pr_c2, inp, H, inp_bf, H, M, H, D4);

    // GRU
    GEMM(0, inp_bf, H, Wih, bih, gi, H3, nullptr, 0, M, H3, H);
    GEMM(0, st_bf,  H, Whh, bhh, gh, H3, nullptr, 0, B, H3, H);
    q2a_gru<<<EW(M * H), 256, 0, stream>>>(gi, gh, st, states, A, H, M * H);

    // projection head + loss
    LN(states, pj_g1, pj_b1, lnbuf, M, H);
    GEMM(1, lnbuf, H, Wpj1, pj_c1, projh, H, nullptr, 0, M, H, H);
    q2a_proj_ln_dot<<<M, 256, 0, stream>>>(projh, pj_g2, pj_b2, pj_w2, pj_c2, logits, H);
    q2a_loss<<<B, 32, 0, stream>>>(logits, labels, s, S, A, scale, (float*)d_out);

    // next state = inp[label]
    q2a_state_gather<<<EW(B * H), 256, 0, stream>>>(inp, labels, s, S, A, H, st, st_bf, B * H);
  }
}
